// TaskAwarePromptPool_2637109920224
// MI455X (gfx1250) — matrix operand order repaired and checked by reading, not compiled, not run on hardware
//
#include <hip/hip_runtime.h>
#include <hip/hip_bf16.h>

typedef _Float16 f16;
typedef __attribute__((ext_vector_type(16))) _Float16 v16h;
typedef __attribute__((ext_vector_type(8)))  _Float16 half8;
typedef __attribute__((ext_vector_type(8)))  float    v8f;
typedef __attribute__((ext_vector_type(4)))  unsigned int u32x4;
typedef __attribute__((ext_vector_type(4)))  int          i32x4;
typedef __attribute__((ext_vector_type(8)))  int          i32x8;

#define NT 10      // T tasks
#define BQ 128     // batch
#define DIN 768
#define HID 384
#define EMB 768
#define GOUT 6144  // PLEN*EMB
#define TOPK 3

// ---------------------------------------------------------------------------
// fp32 (T,K,N) -> f16 transposed (T,N,K): makes WMMA B-operand loads b128s.
// ---------------------------------------------------------------------------
__global__ void cvt_transpose_f16(const float* __restrict__ src, f16* __restrict__ dst,
                                  int T, int K, int N) {
    long long i = (long long)blockIdx.x * blockDim.x + threadIdx.x;
    long long tot = (long long)T * K * N;
    if (i >= tot) return;
    int n = (int)(i % N);
    long long r = i / N;
    int k = (int)(r % K);
    int t = (int)(r / K);
    dst[(long long)t * N * K + (long long)n * K + k] = (f16)src[i];
}

// ---------------------------------------------------------------------------
// Generic task-batched WMMA GEMM:  C[z] = act(A[z] (MxK) @ Bt[z]^T + bias[z])
// 32x32 output tile per wave (2x2 of 16x16 WMMA): 4 WMMAs per 8 b128 loads,
// double the register-level reuse of a 16x16 tile. 4 waves / block, grid.y=T.
// ---------------------------------------------------------------------------
__global__ __launch_bounds__(128)
void gemm_wmma_f16(const f16* __restrict__ A, long long sA, int lda,
                   const f16* __restrict__ Bt, long long sB, int ldb,
                   const float* __restrict__ bias, long long sBias,
                   f16* outH, long long sCh,
                   f16* outHT, long long sChT,
                   float* outF, long long sCf,
                   int M, int N, int K, int relu) {
    const int lane = threadIdx.x & 31;
    const int wid  = threadIdx.x >> 5;
    const int ntiles = N >> 5;                 // 32-wide tiles
    const int mtiles = M >> 5;
    const int w = blockIdx.x * 4 + wid;
    if (w >= ntiles * mtiles) return;          // uniform per wave: EXEC stays all-ones
    const int mt = w / ntiles, nt = w % ntiles;
    const int z = blockIdx.y;
    A  += (long long)z * sA;
    Bt += (long long)z * sB;

    const int l15 = lane & 15;
    const int hi  = lane >> 4;                  // lane group 0/1
    const int mbase = mt << 5, nbase = nt << 5;

    // A frag: lane = row, halves {0..7}=K{hi*8..}, {8..15}=K{hi*8+16..}
    const f16* aptr0 = A + (long long)(mbase + l15) * lda + hi * 8;
    const f16* aptr1 = aptr0 + (long long)16 * lda;
    // B frag: lane = col, halves {0..7}=K{hi*8..}, {8..15}=K{hi*8+16..}
    const f16* bptr0 = Bt + (long long)(nbase + l15) * ldb + hi * 8;
    const f16* bptr1 = bptr0 + (long long)16 * ldb;

    union U16 { v16h v; half8 h[2]; };
    v8f acc00 = {}, acc01 = {}, acc10 = {}, acc11 = {};
    for (int k0 = 0; k0 < K; k0 += 32) {
        U16 a0, a1, b0, b1;
        a0.h[0] = *(const half8*)(aptr0 + k0);
        a0.h[1] = *(const half8*)(aptr0 + k0 + 16);
        a1.h[0] = *(const half8*)(aptr1 + k0);
        a1.h[1] = *(const half8*)(aptr1 + k0 + 16);
        b0.h[0] = *(const half8*)(bptr0 + k0);
        b0.h[1] = *(const half8*)(bptr0 + k0 + 16);
        b1.h[0] = *(const half8*)(bptr1 + k0);
        b1.h[1] = *(const half8*)(bptr1 + k0 + 16);
        acc00 = __builtin_amdgcn_wmma_f32_16x16x32_f16(false, a0.v, false, b0.v, (short)0, acc00, false, false);
        acc01 = __builtin_amdgcn_wmma_f32_16x16x32_f16(false, a0.v, false, b1.v, (short)0, acc01, false, false);
        acc10 = __builtin_amdgcn_wmma_f32_16x16x32_f16(false, a1.v, false, b0.v, (short)0, acc10, false, false);
        acc11 = __builtin_amdgcn_wmma_f32_16x16x32_f16(false, a1.v, false, b1.v, (short)0, acc11, false, false);
    }

    auto store = [&](const v8f& acc, int mb, int nb) {
        const int n = nb + l15;
        const float bv = bias ? bias[(long long)z * sBias + n] : 0.0f;
        const int m0 = mb + hi * 8;             // C/D layout: VGPR r -> row r + 8*hi
#pragma unroll
        for (int r = 0; r < 8; ++r) {
            const int m = m0 + r;
            float v = acc[r] + bv;
            if (relu) v = v > 0.f ? v : 0.f;
            if (outH)  outH [(long long)z * sCh  + (long long)m * N + n] = (f16)v;
            if (outHT) outHT[(long long)z * sChT + (long long)n * M + m] = (f16)v;
            if (outF)  outF [(long long)z * sCf  + (long long)m * N + n] = v;
        }
    };
    store(acc00, mbase,      nbase);
    store(acc01, mbase,      nbase + 16);
    store(acc10, mbase + 16, nbase);
    store(acc11, mbase + 16, nbase + 16);
}

// ---------------------------------------------------------------------------
// TDM-staged GEMM for the big weight stream (K=384 fixed, N multiple of 64).
// Block = 256 threads (8 waves) computes a 128x64 region. Wave 0 issues one
// tensor_load_to_lds pulling the 64x384 f16 B-tile (48KB) into LDS; all waves
// then read B fragments with ds_load_b128 (4x reuse across the 4 M-tiles)
// while A streams from L2.
// ---------------------------------------------------------------------------
#define KTDM 384
__global__ __launch_bounds__(256)
void gemm_wmma_tdm(const f16* __restrict__ A, long long sA,
                   const f16* __restrict__ Bt, long long sB,
                   const float* __restrict__ bias, long long sBias,
                   float* __restrict__ outF, long long sCf,
                   int N, int relu) {
    __shared__ f16 Bs[64 * KTDM];               // 48 KB of the WGP's 320 KB LDS
    const int lane = threadIdx.x & 31;
    const int wid  = threadIdx.x >> 5;
    const int z = blockIdx.y;
    const int nblk = blockIdx.x * 64;

    // ---- wave 0: build D# in scalar regs and issue the TDM load ----
    if (wid == 0) {
        const unsigned long long gaddr =
            (unsigned long long)(const char*)(Bt + (long long)z * sB + (long long)nblk * KTDM);
        const unsigned ldsaddr = (unsigned)(unsigned long long)(const char*)Bs; // LDS = addr[31:0]
        u32x4 g0;
        g0.x = 1u;                               // count=1, is_restore=0, gather off
        g0.y = ldsaddr;                          // lds_addr
        g0.z = (unsigned)(gaddr & 0xffffffffull);
        g0.w = (unsigned)((gaddr >> 32) & 0x1ffffffull) | (2u << 30); // type=2 (image)
        i32x8 g1;
        g1[0] = (1 << 16);                       // workgroup_mask=0, data_size=1 (2 bytes)
        g1[1] = (KTDM & 0xffff) << 16;           // tensor_dim0[15:0]
        g1[2] = ((KTDM >> 16) & 0xffff) | ((64 & 0xffff) << 16); // dim0 hi | tensor_dim1 lo
        g1[3] = ((64 >> 16) & 0xffff) | ((KTDM & 0xffff) << 16); // dim1 hi | tile_dim0
        g1[4] = 64;                              // tile_dim1 = 64, tile_dim2 = 0
        g1[5] = KTDM;                            // tensor_dim0_stride[31:0]
        g1[6] = 0;                               // stride hi | tensor_dim1_stride lo
        g1[7] = 0;
        i32x4 gz = {0, 0, 0, 0};
#if defined(__clang_major__) && (__clang_major__ >= 23)
        i32x8 gz8 = {0, 0, 0, 0, 0, 0, 0, 0};
        __builtin_amdgcn_tensor_load_to_lds(g0, g1, gz, gz, gz8, 0);
#else
        __builtin_amdgcn_tensor_load_to_lds(g0, g1, gz, gz, 0);
#endif
        __builtin_amdgcn_s_wait_tensorcnt(0);
    }
    __syncthreads();

    // ---- 8 waves: wave w -> M-tile (w>>1)*32, N-subtile (w&1)*32 ----
    const int l15 = lane & 15;
    const int hi  = lane >> 4;
    const int mb  = (wid >> 1) * 32;
    const int nb  = (wid & 1) * 32;

    const f16* aptr0 = A + (long long)z * sA + (long long)(mb + l15) * KTDM + hi * 8;
    const f16* aptr1 = aptr0 + (long long)16 * KTDM;
    const f16* bptr0 = Bs + (nb + l15) * KTDM + hi * 8;
    const f16* bptr1 = bptr0 + 16 * KTDM;

    union U16 { v16h v; half8 h[2]; };
    v8f acc00 = {}, acc01 = {}, acc10 = {}, acc11 = {};
#pragma unroll 4
    for (int k0 = 0; k0 < KTDM; k0 += 32) {
        U16 a0, a1, b0, b1;
        a0.h[0] = *(const half8*)(aptr0 + k0);
        a0.h[1] = *(const half8*)(aptr0 + k0 + 16);
        a1.h[0] = *(const half8*)(aptr1 + k0);
        a1.h[1] = *(const half8*)(aptr1 + k0 + 16);
        b0.h[0] = *(const half8*)(bptr0 + k0);
        b0.h[1] = *(const half8*)(bptr0 + k0 + 16);
        b1.h[0] = *(const half8*)(bptr1 + k0);
        b1.h[1] = *(const half8*)(bptr1 + k0 + 16);
        acc00 = __builtin_amdgcn_wmma_f32_16x16x32_f16(false, a0.v, false, b0.v, (short)0, acc00, false, false);
        acc01 = __builtin_amdgcn_wmma_f32_16x16x32_f16(false, a0.v, false, b1.v, (short)0, acc01, false, false);
        acc10 = __builtin_amdgcn_wmma_f32_16x16x32_f16(false, a1.v, false, b0.v, (short)0, acc10, false, false);
        acc11 = __builtin_amdgcn_wmma_f32_16x16x32_f16(false, a1.v, false, b1.v, (short)0, acc11, false, false);
    }

    auto store = [&](const v8f& acc, int m_base, int n_loc) {
        const int n = nblk + n_loc + l15;
        const float bv = bias ? bias[(long long)z * sBias + n] : 0.0f;
        const int m0 = m_base + hi * 8;
#pragma unroll
        for (int r = 0; r < 8; ++r) {
            float v = acc[r] + bv;
            if (relu) v = v > 0.f ? v : 0.f;
            outF[(long long)z * sCf + (long long)(m0 + r) * N + n] = v;
        }
    };
    store(acc00, mb,      nb);
    store(acc01, mb,      nb + 16);
    store(acc10, mb + 16, nb);
    store(acc11, mb + 16, nb + 16);
}

// ---------------------------------------------------------------------------
// Row-wise L2 normalize (optionally emit f16 copy). One block per row.
// ---------------------------------------------------------------------------
__global__ __launch_bounds__(256)
void l2rows(const float* x, float* o, f16* oh, int D) {
    __shared__ float red[256];
    const long long row = blockIdx.x;
    const float* r = x + row * D;
    float s = 0.f;
    for (int i = threadIdx.x; i < D; i += 256) { float v = r[i]; s += v * v; }
    red[threadIdx.x] = s; __syncthreads();
    for (int st = 128; st > 0; st >>= 1) {
        if (threadIdx.x < st) red[threadIdx.x] += red[threadIdx.x + st];
        __syncthreads();
    }
    const float inv = 1.0f / fmaxf(sqrtf(red[0]), 1e-12f);
    for (int i = threadIdx.x; i < D; i += 256) {
        float v = r[i] * inv;
        o[row * D + i] = v;
        if (oh) oh[row * D + i] = (f16)v;
    }
}

// 128-wide attention softmax: P = softmax(S * scale) per row, f16 out.
__global__ __launch_bounds__(128)
void softmax128(const float* __restrict__ S, f16* __restrict__ P, float scale) {
    __shared__ float red[128];
    const long long row = blockIdx.x;
    const int j = threadIdx.x;
    const float v = S[row * 128 + j] * scale;
    red[j] = v; __syncthreads();
    for (int st = 64; st > 0; st >>= 1) { if (j < st) red[j] = fmaxf(red[j], red[j + st]); __syncthreads(); }
    const float m = red[0]; __syncthreads();
    const float e = expf(v - m);
    red[j] = e; __syncthreads();
    for (int st = 64; st > 0; st >>= 1) { if (j < st) red[j] += red[j + st]; __syncthreads(); }
    P[row * 128 + j] = (f16)(e / red[0]);
}

__global__ void add_f16(const f16* a, const f16* b, f16* c, long long n) {
    long long i = (long long)blockIdx.x * blockDim.x + threadIdx.x;
    if (i < n) c[i] = (f16)((float)a[i] + (float)b[i]);
}

// sim[b][t] = dot(qn[b], keys[t][b]) over EMB. One 64-thread block per (b,t).
__global__ __launch_bounds__(64)
void sim_dot(const float* __restrict__ qn, const float* __restrict__ keys,
             float* __restrict__ sim, int B, int T, int D) {
    __shared__ float red[64];
    const int bt = blockIdx.x, b = bt / T, t = bt % T;
    const float* kr = keys + ((long long)t * B + b) * D;
    const float* qr = qn + (long long)b * D;
    float s = 0.f;
    for (int i = threadIdx.x; i < D; i += 64) s += qr[i] * kr[i];
    red[threadIdx.x] = s; __syncthreads();
    for (int st = 32; st > 0; st >>= 1) { if (threadIdx.x < st) red[threadIdx.x] += red[threadIdx.x + st]; __syncthreads(); }
    if (threadIdx.x == 0) sim[b * T + t] = red[0];
}

// Routing: softmax(sim/TEMP), top-3, conf/entropy gate, collapse blend+weights
// into 3 gather coefficients per row.
__global__ __launch_bounds__(128)
void route(const float* __restrict__ sim, int* __restrict__ idx,
           float* __restrict__ coef, int B, int T) {
    const int b = blockIdx.x * blockDim.x + threadIdx.x;
    if (b >= B) return;
    float p[16];
    float mx = -1e30f;
    for (int t = 0; t < T; ++t) { float s = sim[b * T + t] * 5.0f; /* /0.2 */ p[t] = s; mx = fmaxf(mx, s); }
    float sum = 0.f;
    for (int t = 0; t < T; ++t) { p[t] = expf(p[t] - mx); sum += p[t]; }
    for (int t = 0; t < T; ++t) p[t] /= sum;
    int id[TOPK]; float tv[TOPK]; bool used[16] = {};
    for (int k = 0; k < TOPK; ++k) {
        int best = -1; float bv = -1e30f;
        for (int t = 0; t < T; ++t) if (!used[t] && p[t] > bv) { bv = p[t]; best = t; }
        used[best] = true; id[k] = best; tv[k] = bv;
    }
    float ent = 0.f;
    for (int t = 0; t < T; ++t) ent -= p[t] * logf(p[t] + 1e-12f);
    ent /= logf((float)T);
    const float conf = tv[0];
    const float blend = (conf < 0.35f || ent > 0.75f) ? 0.5f : 0.0f;
    float ws = fmaxf(tv[0] + tv[1] + tv[2], 1e-12f);
    coef[b * 3 + 0] = (1.0f - blend) + blend * (tv[0] / ws);
    coef[b * 3 + 1] = blend * (tv[1] / ws);
    coef[b * 3 + 2] = blend * (tv[2] / ws);
    idx[b * 3 + 0] = id[0]; idx[b * 3 + 1] = id[1]; idx[b * 3 + 2] = id[2];
}

// fused[b][j] = sum_k coef[b][k] * Pgen[idx[b][k]][b][j]; scatter to stacked
// output layout (2, B, PLEN/2, EMB).
__global__ void combine(const float* __restrict__ Pgen, const int* __restrict__ idx,
                        const float* __restrict__ coef, float* __restrict__ out,
                        int B, int J) {
    long long i = (long long)blockIdx.x * blockDim.x + threadIdx.x;
    if (i >= (long long)B * J) return;
    const int j = (int)(i % J);
    const int b = (int)(i / J);
    float acc = 0.f;
#pragma unroll
    for (int k = 0; k < TOPK; ++k) {
        const int t = idx[b * 3 + k];
        acc += coef[b * 3 + k] * Pgen[((long long)t * B + b) * J + j];
    }
    const int l = j / EMB, e = j % EMB;
    const int s = l >> 2, l2 = l & 3;
    out[(((long long)s * B + b) * 4 + l2) * EMB + e] = acc;
}

// ---------------------------------------------------------------------------
extern "C" void kernel_launch(void* const* d_in, const int* in_sizes, int n_in,
                              void* d_out, int out_size, void* d_ws, size_t ws_size,
                              hipStream_t stream) {
    // --- input pointers (setup_inputs dict order) ---
    const float* x_query = (const float*)d_in[0];
    const float* gW1 = (const float*)d_in[3];   // (T,768,384)
    const float* gb1 = (const float*)d_in[4];
    const float* gW2 = (const float*)d_in[5];   // (T,384,6144)
    const float* gb2 = (const float*)d_in[6];
    const float* lin1W[3] = {(const float*)d_in[7],  (const float*)d_in[19], (const float*)d_in[29]};
    const float* lin1b[3] = {(const float*)d_in[8],  (const float*)d_in[20], (const float*)d_in[30]};
    const float* qW[3]    = {(const float*)d_in[9],  (const float*)d_in[21], (const float*)d_in[31]};
    const float* qb[3]    = {(const float*)d_in[10], (const float*)d_in[22], (const float*)d_in[32]};
    const float* kW[3]    = {(const float*)d_in[11], (const float*)d_in[23], (const float*)d_in[33]};
    const float* kb[3]    = {(const float*)d_in[12], (const float*)d_in[24], (const float*)d_in[34]};
    const float* vW[3]    = {(const float*)d_in[13], (const float*)d_in[25], (const float*)d_in[35]};
    const float* vb[3]    = {(const float*)d_in[14], (const float*)d_in[26], (const float*)d_in[36]};
    const float* l2W[3]   = {(const float*)d_in[15], (const float*)d_in[27], (const float*)d_in[37]};
    const float* l2b[3]   = {(const float*)d_in[16], (const float*)d_in[28], (const float*)d_in[38]};
    const float* scW = (const float*)d_in[17];
    const float* scb = (const float*)d_in[18];
    const float* outW = (const float*)d_in[39]; // (T,384,768)
    const float* outb = (const float*)d_in[40];

    // --- workspace layout (256B aligned bump allocator) ---
    char* base = (char*)d_ws;
    size_t off = 0;
    auto alloc = [&](size_t bytes) -> void* {
        void* p = base + off;
        off = (off + bytes + 255) & ~(size_t)255;
        return p;
    };
    float* qn    = (float*)alloc((size_t)BQ * DIN * 4);
    f16*  qn_h   = (f16*)  alloc((size_t)BQ * DIN * 2);
    f16*  W1t    = (f16*)  alloc((size_t)NT * HID * DIN * 2);
    f16*  W2t    = (f16*)  alloc((size_t)NT * GOUT * HID * 2);
    f16*  outWt  = (f16*)  alloc((size_t)NT * EMB * HID * 2);
    f16*  lin1t[3], *qt[3], *kt[3], *vt[3], *l2t[3];
    for (int i = 0; i < 3; ++i) {
        const int din = (i == 0) ? DIN : HID;
        lin1t[i] = (f16*)alloc((size_t)NT * HID * din * 2);
        qt[i]    = (f16*)alloc((size_t)NT * HID * HID * 2);
        kt[i]    = (f16*)alloc((size_t)NT * HID * HID * 2);
        vt[i]    = (f16*)alloc((size_t)NT * HID * HID * 2);
        l2t[i]   = (f16*)alloc((size_t)NT * HID * HID * 2);
    }
    f16*  sct   = (f16*)alloc((size_t)NT * HID * DIN * 2);
    f16*  h_h   = (f16*)alloc((size_t)NT * BQ * HID * 2);
    f16*  q_h   = (f16*)alloc((size_t)NT * BQ * HID * 2);
    f16*  k_h   = (f16*)alloc((size_t)NT * BQ * HID * 2);
    f16*  vT_h  = (f16*)alloc((size_t)NT * HID * BQ * 2);
    float* S    = (float*)alloc((size_t)NT * BQ * BQ * 4);
    f16*  P_h   = (f16*)alloc((size_t)NT * BQ * BQ * 2);
    f16*  a_h   = (f16*)alloc((size_t)NT * BQ * HID * 2);
    f16*  sc_h  = (f16*)alloc((size_t)NT * BQ * HID * 2);
    f16*  asum  = (f16*)alloc((size_t)NT * BQ * HID * 2);
    f16*  x_h   = (f16*)alloc((size_t)NT * BQ * HID * 2);
    float* y    = (float*)alloc((size_t)NT * BQ * EMB * 4);
    float* simb = (float*)alloc((size_t)BQ * NT * 4);
    int*   idxb = (int*)  alloc((size_t)BQ * 3 * 4);
    float* coefb= (float*)alloc((size_t)BQ * 3 * 4);
    f16*  hgen  = (f16*)alloc((size_t)NT * BQ * HID * 2);
    float* Pgen = (float*)alloc((size_t)NT * BQ * GOUT * 4);
    (void)ws_size; (void)in_sizes; (void)n_in; (void)out_size;

    auto cvt = [&](const float* src, f16* dst, int T, int K, int N) {
        long long tot = (long long)T * K * N;
        cvt_transpose_f16<<<dim3((unsigned)((tot + 255) / 256)), 256, 0, stream>>>(src, dst, T, K, N);
    };
    auto gemm = [&](const f16* A, long long sA, int lda,
                    const f16* Bt, long long sB, int ldb,
                    const float* bias, long long sBias,
                    f16* oh, long long sh, f16* oht, long long sht,
                    float* of, long long sf,
                    int M, int N, int K, int relu, int T) {
        int tiles = (M / 32) * (N / 32);
        gemm_wmma_f16<<<dim3((tiles + 3) / 4, T), 128, 0, stream>>>(
            A, sA, lda, Bt, sB, ldb, bias, sBias, oh, sh, oht, sht, of, sf, M, N, K, relu);
    };

    // --- phase 0: weight fp32 -> f16^T (one pass, halves HBM/L2 traffic) ---
    cvt(gW1, W1t, NT, DIN, HID);
    cvt(gW2, W2t, NT, HID, GOUT);
    cvt(outW, outWt, NT, HID, EMB);
    cvt(scW, sct, NT, DIN, HID);
    for (int i = 0; i < 3; ++i) {
        const int din = (i == 0) ? DIN : HID;
        cvt(lin1W[i], lin1t[i], NT, din, HID);
        cvt(qW[i], qt[i], NT, HID, HID);
        cvt(kW[i], kt[i], NT, HID, HID);
        cvt(vW[i], vt[i], NT, HID, HID);
        cvt(l2W[i], l2t[i], NT, HID, HID);
    }

    // --- phase 1: normalize query ---
    l2rows<<<BQ, 256, 0, stream>>>(x_query, qn, qn_h, DIN);

    // --- phase 2: key-learner towers, all T tasks batched via grid.y ---
    const long long sAct = (long long)BQ * HID;    // per-task activation stride
    const long long sQKV = (long long)HID * HID;
    for (int i = 0; i < 3; ++i) {
        const int din = (i == 0) ? DIN : HID;
        const f16* Ain = (i == 0) ? qn_h : x_h;
        const long long sAin = (i == 0) ? 0 : sAct;
        // h = relu(x @ lin1 + b)
        gemm(Ain, sAin, din, lin1t[i], (long long)HID * din, din, lin1b[i], HID,
             h_h, sAct, nullptr, 0, nullptr, 0, BQ, HID, din, 1, NT);
        // q, k (row-major f16), v (transposed for attention B-operand)
        gemm(h_h, sAct, HID, qt[i], sQKV, HID, qb[i], HID,
             q_h, sAct, nullptr, 0, nullptr, 0, BQ, HID, HID, 0, NT);
        gemm(h_h, sAct, HID, kt[i], sQKV, HID, kb[i], HID,
             k_h, sAct, nullptr, 0, nullptr, 0, BQ, HID, HID, 0, NT);
        gemm(h_h, sAct, HID, vt[i], sQKV, HID, vb[i], HID,
             nullptr, 0, vT_h, sAct, nullptr, 0, BQ, HID, HID, 0, NT);
        // S = q @ k^T  (k row-major IS the N x K operand)
        gemm(q_h, sAct, HID, k_h, sAct, HID, nullptr, 0,
             nullptr, 0, nullptr, 0, S, (long long)BQ * BQ, BQ, BQ, HID, 0, NT);
        softmax128<<<NT * BQ, 128, 0, stream>>>(S, P_h, 0.051031036307982884f /*1/sqrt(384)*/);
        // a = P @ v   (vT is the N x K operand)
        gemm(P_h, (long long)BQ * BQ, BQ, vT_h, sAct, BQ, nullptr, 0,
             a_h, sAct, nullptr, 0, nullptr, 0, BQ, HID, BQ, 0, NT);
        // residual path
        if (i == 0) {
            gemm(qn_h, 0, DIN, sct, (long long)HID * DIN, DIN, scb, HID,
                 sc_h, sAct, nullptr, 0, nullptr, 0, BQ, HID, DIN, 0, NT);
            add_f16<<<(NT * BQ * HID + 255) / 256, 256, 0, stream>>>(a_h, sc_h, asum, (long long)NT * BQ * HID);
        } else {
            add_f16<<<(NT * BQ * HID + 255) / 256, 256, 0, stream>>>(a_h, x_h, asum, (long long)NT * BQ * HID);
        }
        // x = relu((a + sc) @ lin2 + b)
        gemm(asum, sAct, HID, l2t[i], sQKV, HID, l2b[i], HID,
             x_h, sAct, nullptr, 0, nullptr, 0, BQ, HID, HID, 1, NT);
    }
    // y = relu(x) @ out_W + out_b   (x already >= 0), then L2 normalize
    gemm(x_h, sAct, HID, outWt, (long long)EMB * HID, HID, outb, EMB,
         nullptr, 0, nullptr, 0, y, (long long)BQ * EMB, BQ, EMB, HID, 0, NT);
    l2rows<<<NT * BQ, 256, 0, stream>>>(y, y, nullptr, EMB);

    // --- phase 3: routing ---
    sim_dot<<<BQ * NT, 64, 0, stream>>>(qn, y, simb, BQ, NT, EMB);
    route<<<1, 128, 0, stream>>>(simb, idxb, coefb, BQ, NT);

    // --- phase 4: generator ---
    gemm(qn_h, 0, DIN, W1t, (long long)HID * DIN, DIN, gb1, HID,
         hgen, sAct, nullptr, 0, nullptr, 0, BQ, HID, DIN, 1, NT);
    // big weight stream (W2t ~47MB f16): TDM-staged LDS GEMM, 128x64 per block
    gemm_wmma_tdm<<<dim3(GOUT / 64, NT), 256, 0, stream>>>(
        hgen, sAct, W2t, (long long)GOUT * HID, gb2, GOUT,
        Pgen, (long long)BQ * GOUT, GOUT, 1);

    // --- phase 5: top-k blend + stacked output ---
    combine<<<(BQ * GOUT + 255) / 256, 256, 0, stream>>>(Pgen, idxb, coefb, (float*)d_out, BQ, GOUT);
}
